// LLMSASrec_33406255628497
// MI455X (gfx1250) — matrix operand order repaired and checked by reading, not compiled, not run on hardware
//
#include <hip/hip_runtime.h>
#include <cstddef>
#include <cstdint>

// ---------------- types ----------------
typedef __bf16 bf16_t;
typedef __attribute__((ext_vector_type(16))) __bf16 v16bf;
typedef __attribute__((ext_vector_type(8)))  float  v8f;

#define Bsz 256
#define Lsz 201
#define Dsz 1536
#define Hsz 64
#define FFsz 256
#define BL 51456          // Bsz*Lsz
#define LPADK 224         // Lsz padded to 14*16

// output offsets (floats) in d_out, reference return order
#define O0 0ull                    // prec_vec[:, :-1, :]   256*200*64
#define O1 3276800ull              // target_pos_embs       256*200*64
#define O2 6553600ull              // target_neg_embs       256*200*64
#define O3 9830400ull              // concat(embs, prec)    256*201*128
#define OSIM 16416768ull           // similarity scalar

// ---------------- helpers ----------------
__device__ inline v8f v8f_zero() {
  v8f z = {0.f,0.f,0.f,0.f,0.f,0.f,0.f,0.f};
  return z;
}

__device__ inline v8f wmma_bf16(v16bf a, v16bf b, v8f c) {
  return __builtin_amdgcn_wmma_f32_16x16x32_bf16(false, a, false, b, (short)0, c, false, false);
}

// A fragment (16x32 bf16) from fp32 row-major source.
// Layout used here: lane half h: element e -> K = kb+16h+e (contiguous)
__device__ inline v16bf load_a_f32(const float* __restrict__ row, int kb, int h) {
  const float* p0 = row + kb + 16*h;
  const float* p1 = p0 + 8;
  float4 x0 = *(const float4*)p0;
  float4 x1 = *(const float4*)(p0+4);
  float4 y0 = *(const float4*)p1;
  float4 y1 = *(const float4*)(p1+4);
  v16bf a;
  a[0]=(bf16_t)x0.x; a[1]=(bf16_t)x0.y; a[2]=(bf16_t)x0.z; a[3]=(bf16_t)x0.w;
  a[4]=(bf16_t)x1.x; a[5]=(bf16_t)x1.y; a[6]=(bf16_t)x1.z; a[7]=(bf16_t)x1.w;
  a[8]=(bf16_t)y0.x; a[9]=(bf16_t)y0.y; a[10]=(bf16_t)y0.z; a[11]=(bf16_t)y0.w;
  a[12]=(bf16_t)y1.x; a[13]=(bf16_t)y1.y; a[14]=(bf16_t)y1.z; a[15]=(bf16_t)y1.w;
  return a;
}

// A fragment from bf16 row (LDS or global)
__device__ inline v16bf load_a_bf16(const bf16_t* row, int kb, int h) {
  const bf16_t* p0 = row + kb + 16*h;
  const bf16_t* p1 = p0 + 8;
  v16bf a;
#pragma unroll
  for (int e = 0; e < 8; ++e) { a[e] = p0[e]; a[8+e] = p1[e]; }
  return a;
}

// B fragment (32x16) from pre-transposed weights WT[N][K] (bf16).
// Layout used here: b[e] = W[kb+16h+e][col] = WT[col*K + kb+16h+e]  (contiguous)
__device__ inline v16bf load_b_glb(const bf16_t* __restrict__ WT, int K, int col, int kb, int h) {
  return *(const v16bf*)(WT + (size_t)col*K + kb + 16*h);
}

__device__ inline float gelu_tanh(float x) {
  float x3 = x*x*x;
  return 0.5f*x*(1.f + tanhf(0.7978845608028654f*(x + 0.044715f*x3)));
}

// ---------------- weight convert + transpose: in[K][N] f32 -> out[N][K] bf16 ----
__global__ void cvt_transpose_kernel(const float* __restrict__ in, bf16_t* __restrict__ out,
                                     int K, int N) {
  int idx = blockIdx.x*256 + threadIdx.x;
  if (idx >= K*N) return;
  int k = idx / N, n = idx % N;
  out[(size_t)n*K + k] = (bf16_t)in[idx];
}

// ---------------- fused  gelu((X@fc_w+b)@mlp_w+b) ----------------
// X: (BL x 1536) f32, out: (BL x 64) f32. grid 402 x 256thr (8 waves x 16 rows)
__global__ __launch_bounds__(256) void fc_mlp_kernel(
    const float* __restrict__ X, const bf16_t* __restrict__ fcwT,
    const float* __restrict__ fcb, const bf16_t* __restrict__ mlpwT,
    const float* __restrict__ mlpb, float* __restrict__ out) {
  __shared__ __attribute__((aligned(32))) bf16_t Tls[8][16][72];
  const int wave = threadIdx.x >> 5, lane = threadIdx.x & 31;
  const int h = lane >> 4, n16 = lane & 15;
  const int row0 = blockIdx.x*128 + wave*16;
  const float* Arow = X + (size_t)(row0 + n16)*Dsz;
  v8f c0=v8f_zero(), c1=v8f_zero(), c2=v8f_zero(), c3=v8f_zero();
#pragma unroll 2
  for (int kb = 0; kb < Dsz; kb += 32) {
    __builtin_prefetch(Arow + kb + 512, 0, 0);   // stream ~2KB ahead -> global_prefetch_b8
    v16bf a  = load_a_f32(Arow, kb, h);
    v16bf b0 = load_b_glb(fcwT, Dsz,      n16, kb, h);
    v16bf b1 = load_b_glb(fcwT, Dsz, 16 + n16, kb, h);
    v16bf b2 = load_b_glb(fcwT, Dsz, 32 + n16, kb, h);
    v16bf b3 = load_b_glb(fcwT, Dsz, 48 + n16, kb, h);
    c0 = wmma_bf16(a, b0, c0); c1 = wmma_bf16(a, b1, c1);
    c2 = wmma_bf16(a, b2, c2); c3 = wmma_bf16(a, b3, c3);
  }
#pragma unroll
  for (int r = 0; r < 8; ++r) {
    int m = r + 8*h;
    Tls[wave][m][n16]      = (bf16_t)(c0[r] + fcb[n16]);
    Tls[wave][m][16 + n16] = (bf16_t)(c1[r] + fcb[16 + n16]);
    Tls[wave][m][32 + n16] = (bf16_t)(c2[r] + fcb[32 + n16]);
    Tls[wave][m][48 + n16] = (bf16_t)(c3[r] + fcb[48 + n16]);
  }
  asm volatile("s_wait_dscnt 0" ::: "memory");   // wave-local LDS handoff
  v8f d0=v8f_zero(), d1=v8f_zero(), d2=v8f_zero(), d3=v8f_zero();
#pragma unroll
  for (int kb = 0; kb < 64; kb += 32) {
    v16bf a  = load_a_bf16(&Tls[wave][n16][0], kb, h);
    v16bf b0 = load_b_glb(mlpwT, 64,      n16, kb, h);
    v16bf b1 = load_b_glb(mlpwT, 64, 16 + n16, kb, h);
    v16bf b2 = load_b_glb(mlpwT, 64, 32 + n16, kb, h);
    v16bf b3 = load_b_glb(mlpwT, 64, 48 + n16, kb, h);
    d0 = wmma_bf16(a, b0, d0); d1 = wmma_bf16(a, b1, d1);
    d2 = wmma_bf16(a, b2, d2); d3 = wmma_bf16(a, b3, d3);
  }
#pragma unroll
  for (int r = 0; r < 8; ++r) {
    float* op = out + (size_t)(row0 + r + 8*h)*64;
    op[n16]      = gelu_tanh(d0[r] + mlpb[n16]);
    op[16 + n16] = gelu_tanh(d1[r] + mlpb[16 + n16]);
    op[32 + n16] = gelu_tanh(d2[r] + mlpb[32 + n16]);
    op[48 + n16] = gelu_tanh(d3[r] + mlpb[48 + n16]);
  }
}

// ---------------- generic WMMA GEMM: C = act(A@W + bias) ----------------
// A: (BL x K) f32, WT: (N x K) bf16, C: (BL x N) f32. grid (402, N/64)
template<int K, int ACT>
__global__ __launch_bounds__(256) void gemm_wmma_kernel(
    const float* __restrict__ A, const bf16_t* __restrict__ WT,
    const float* __restrict__ bias, float* __restrict__ Cout, const int N) {
  const int wave = threadIdx.x >> 5, lane = threadIdx.x & 31;
  const int h = lane >> 4, n16 = lane & 15;
  const int row0 = blockIdx.x*128 + wave*16;
  const int col0 = blockIdx.y*64;
  const float* Arow = A + (size_t)(row0 + n16)*K;
  v8f c0=v8f_zero(), c1=v8f_zero(), c2=v8f_zero(), c3=v8f_zero();
#pragma unroll
  for (int kb = 0; kb < K; kb += 32) {
    v16bf a  = load_a_f32(Arow, kb, h);
    v16bf b0 = load_b_glb(WT, K, col0 +      n16, kb, h);
    v16bf b1 = load_b_glb(WT, K, col0 + 16 + n16, kb, h);
    v16bf b2 = load_b_glb(WT, K, col0 + 32 + n16, kb, h);
    v16bf b3 = load_b_glb(WT, K, col0 + 48 + n16, kb, h);
    c0 = wmma_bf16(a, b0, c0); c1 = wmma_bf16(a, b1, c1);
    c2 = wmma_bf16(a, b2, c2); c3 = wmma_bf16(a, b3, c3);
  }
#pragma unroll
  for (int r = 0; r < 8; ++r) {
    float* op = Cout + (size_t)(row0 + r + 8*h)*N + col0;
    float v0 = c0[r] + bias[col0 + n16];
    float v1 = c1[r] + bias[col0 + 16 + n16];
    float v2 = c2[r] + bias[col0 + 32 + n16];
    float v3 = c3[r] + bias[col0 + 48 + n16];
    if (ACT) { v0 = gelu_tanh(v0); v1 = gelu_tanh(v1); v2 = gelu_tanh(v2); v3 = gelu_tanh(v3); }
    op[n16] = v0; op[16 + n16] = v1; op[32 + n16] = v2; op[48 + n16] = v3;
  }
}

// Same GEMM but bf16 output (qkv buffer: halves HBM traffic; attention is bf16 anyway)
template<int K>
__global__ __launch_bounds__(256) void gemm_wmma_bf16out_kernel(
    const float* __restrict__ A, const bf16_t* __restrict__ WT,
    const float* __restrict__ bias, bf16_t* __restrict__ Cout, const int N) {
  const int wave = threadIdx.x >> 5, lane = threadIdx.x & 31;
  const int h = lane >> 4, n16 = lane & 15;
  const int row0 = blockIdx.x*128 + wave*16;
  const int col0 = blockIdx.y*64;
  const float* Arow = A + (size_t)(row0 + n16)*K;
  v8f c0=v8f_zero(), c1=v8f_zero(), c2=v8f_zero(), c3=v8f_zero();
#pragma unroll
  for (int kb = 0; kb < K; kb += 32) {
    v16bf a  = load_a_f32(Arow, kb, h);
    v16bf b0 = load_b_glb(WT, K, col0 +      n16, kb, h);
    v16bf b1 = load_b_glb(WT, K, col0 + 16 + n16, kb, h);
    v16bf b2 = load_b_glb(WT, K, col0 + 32 + n16, kb, h);
    v16bf b3 = load_b_glb(WT, K, col0 + 48 + n16, kb, h);
    c0 = wmma_bf16(a, b0, c0); c1 = wmma_bf16(a, b1, c1);
    c2 = wmma_bf16(a, b2, c2); c3 = wmma_bf16(a, b3, c3);
  }
#pragma unroll
  for (int r = 0; r < 8; ++r) {
    bf16_t* op = Cout + (size_t)(row0 + r + 8*h)*N + col0;
    op[n16]      = (bf16_t)(c0[r] + bias[col0 + n16]);
    op[16 + n16] = (bf16_t)(c1[r] + bias[col0 + 16 + n16]);
    op[32 + n16] = (bf16_t)(c2[r] + bias[col0 + 32 + n16]);
    op[48 + n16] = (bf16_t)(c3[r] + bias[col0 + 48 + n16]);
  }
}

// ---------------- flash attention, one (b,head) per block ----------------
// qkv16: (BL x 192) bf16 [q|k|v], out: (BL x 64) f32 (heads concatenated)
__global__ __launch_bounds__(256) void attn_kernel(
    const bf16_t* __restrict__ qkv16, const int* __restrict__ mask,
    float* __restrict__ out) {
  const int b  = blockIdx.x >> 1;
  const int hd = blockIdx.x & 1;
  __shared__ __attribute__((aligned(32))) bf16_t Kls[LPADK*32];   // [key][d]
  __shared__ __attribute__((aligned(32))) bf16_t VT[32*LPADK];    // [d][key]
  __shared__ __attribute__((aligned(32))) bf16_t Pst[8][16][32];  // per-wave P staging
  __shared__ float mstat[LPADK];

  // K tile: async global->LDS copy, 16B per transfer, [key][d] is a straight byte copy.
  {
    const bf16_t* kbase = qkv16 + (size_t)b*Lsz*192 + 64 + hd*32;
    for (int idx = threadIdx.x; idx < Lsz*4; idx += 256) {
      int key = idx >> 2, ch = idx & 3;
      unsigned long long ga = (unsigned long long)(size_t)(kbase + (size_t)key*192 + ch*8);
      unsigned int lo = (unsigned int)(size_t)&Kls[key*32 + ch*8];  // LDS aperture: low 32 bits = LDS addr
      asm volatile("global_load_async_to_lds_b128 %0, %1, off"
                   :: "v"(lo), "v"(ga) : "memory");
    }
  }
  // zero-pad K rows 201..223 (keeps padded scores finite)
  for (int idx = threadIdx.x; idx < (LPADK - Lsz)*32; idx += 256)
    Kls[(Lsz + idx/32)*32 + (idx & 31)] = (bf16_t)0.f;
  // V tile: transposed staging [d][key] (needs shuffle, stays on VALU path)
  for (int idx = threadIdx.x; idx < LPADK*32; idx += 256) {
    int key = idx >> 5, d = idx & 31;
    bf16_t vv = (bf16_t)0.f;
    if (key < Lsz) vv = qkv16[((size_t)(b*Lsz + key))*192 + 128 + hd*32 + d];
    VT[d*LPADK + key] = vv;
  }
  for (int key = threadIdx.x; key < LPADK; key += 256)
    mstat[key] = (key < Lsz) ? (mask[b*Lsz + key] ? 1.f : 0.f) : -1.f;
  asm volatile("s_wait_asynccnt 0" ::: "memory");
  __syncthreads();

  const int wave = threadIdx.x >> 5, lane = threadIdx.x & 31;
  const int h = lane >> 4, n16 = lane & 15;
  const float NEGINF = -__builtin_inff();
  for (int qt = wave; qt < 13; qt += 8) {
    const int qbase = qt*16;
    int lq = qbase + n16; if (lq > 200) lq = 200;
    const bf16_t* qrow = qkv16 + (size_t)(b*Lsz + lq)*192 + hd*32;
    v16bf qa = load_a_bf16(qrow, 0, h);
    v8f o0 = v8f_zero(), o1 = v8f_zero();
    float mrow[8], srow[8];
#pragma unroll
    for (int r = 0; r < 8; ++r) { mrow[r] = NEGINF; srow[r] = 0.f; }

    for (int kb = 0; kb < 7; ++kb) {
      const int k0 = kb*32;
      v16bf kf0 = *(const v16bf*)&Kls[(k0 +      n16)*32 + 16*h];
      v16bf kf1 = *(const v16bf*)&Kls[(k0 + 16 + n16)*32 + 16*h];
      v8f s0 = wmma_bf16(qa, kf0, v8f_zero());
      v8f s1 = wmma_bf16(qa, kf1, v8f_zero());
      const int key0 = k0 + n16, key1 = key0 + 16;
      const float mk0 = mstat[key0], mk1 = mstat[key1];
#pragma unroll
      for (int r = 0; r < 8; ++r) {
        const int l_r = qbase + r + 8*h;
        float a0 = s0[r]*0.17677669529663687f;   // 1/sqrt(32)
        float a1 = s1[r]*0.17677669529663687f;
        a0 += (mk0 < 0.f) ? NEGINF : ((mk0 > 0.f && key0 <= l_r) ? 0.f : -1e9f);
        a1 += (mk1 < 0.f) ? NEGINF : ((mk1 > 0.f && key1 <= l_r) ? 0.f : -1e9f);
        float lm = fmaxf(a0, a1);
        lm = fmaxf(lm, __shfl_xor(lm, 1, 16));
        lm = fmaxf(lm, __shfl_xor(lm, 2, 16));
        lm = fmaxf(lm, __shfl_xor(lm, 4, 16));
        lm = fmaxf(lm, __shfl_xor(lm, 8, 16));
        float mnew  = fmaxf(mrow[r], lm);
        float alpha = __expf(mrow[r] - mnew);
        float p0 = __expf(a0 - mnew), p1 = __expf(a1 - mnew);
        float ps = p0 + p1;
        ps += __shfl_xor(ps, 1, 16); ps += __shfl_xor(ps, 2, 16);
        ps += __shfl_xor(ps, 4, 16); ps += __shfl_xor(ps, 8, 16);
        srow[r] = srow[r]*alpha + ps;
        mrow[r] = mnew;
        o0[r] *= alpha; o1[r] *= alpha;
        Pst[wave][r + 8*h][n16]      = (bf16_t)p0;
        Pst[wave][r + 8*h][16 + n16] = (bf16_t)p1;
      }
      asm volatile("s_wait_dscnt 0" ::: "memory");   // wave-local P transpose handoff
      v16bf pa  = load_a_bf16(&Pst[wave][n16][0], 0, h);
      v16bf vf0 = *(const v16bf*)&VT[(n16)     *LPADK + k0 + 16*h];
      v16bf vf1 = *(const v16bf*)&VT[(16 + n16)*LPADK + k0 + 16*h];
      o0 = wmma_bf16(pa, vf0, o0);
      o1 = wmma_bf16(pa, vf1, o1);
    }
#pragma unroll
    for (int r = 0; r < 8; ++r) {
      int l = qbase + r + 8*h;
      if (l <= 200) {
        float inv = 1.f/srow[r];
        float* op = out + ((size_t)(b*Lsz + l))*64 + hd*32;
        op[n16]      = o0[r]*inv;
        op[16 + n16] = o1[r]*inv;
      }
    }
  }
}

// ---------------- residual + LayerNorm (one wave per 64-col row) ----------------
__global__ __launch_bounds__(256) void resid_ln_kernel(
    const float* __restrict__ x, const float* __restrict__ res, int perRow,
    const float* __restrict__ g, const float* __restrict__ bt, float* __restrict__ out) {
  const int wave = threadIdx.x >> 5, lane = threadIdx.x & 31;
  const int row = blockIdx.x*8 + wave;
  const float* xr = x + (size_t)row*64;
  const float* rr = res + (size_t)(perRow ? row : (row % Lsz))*64;
  const int c0 = lane*2;
  float a0 = xr[c0]   + rr[c0];
  float a1 = xr[c0+1] + rr[c0+1];
  float s = a0 + a1, ss = a0*a0 + a1*a1;
  for (int off = 16; off > 0; off >>= 1) {
    s  += __shfl_xor(s,  off, 32);
    ss += __shfl_xor(ss, off, 32);
  }
  float mean = s*(1.f/64.f);
  float var  = ss*(1.f/64.f) - mean*mean;
  float rinv = rsqrtf(var + 1e-5f);
  float* op = out + (size_t)row*64;
  op[c0]   = (a0 - mean)*rinv*g[c0]   + bt[c0];
  op[c0+1] = (a1 - mean)*rinv*g[c0+1] + bt[c0+1];
}

// ---------------- GNN (rank-1 collapse): masked mean over L ----------------
__global__ __launch_bounds__(256) void gnn_mean_kernel(
    const float* __restrict__ X, const int* __restrict__ mask,
    float* __restrict__ xbar, float* __restrict__ Sv) {
  const int b = blockIdx.x;
  __shared__ float mls[Lsz];
  __shared__ float red[256];
  float ls = 0.f;
  for (int l = threadIdx.x; l < Lsz; l += 256) {
    float m = mask[b*Lsz + l] ? 1.f : 0.f;
    mls[l] = m; ls += m;
  }
  red[threadIdx.x] = ls; __syncthreads();
  for (int o = 128; o > 0; o >>= 1) {
    if (threadIdx.x < o) red[threadIdx.x] += red[threadIdx.x + o];
    __syncthreads();
  }
  const float S = red[0];
  const float inv = (S > 0.f) ? 1.f/S : 0.f;
  for (int d = threadIdx.x; d < Dsz; d += 256) {
    float acc = 0.f;
    for (int l = 0; l < Lsz; ++l) acc += mls[l]*X[((size_t)b*Lsz + l)*Dsz + d];
    xbar[(size_t)b*Dsz + d] = acc*inv;
  }
  if (threadIdx.x == 0) Sv[b] = S;
}

// gvec = relu(xbar@wg1 + bg1) @ wg2  (per-batch matvec chain)
__global__ __launch_bounds__(256) void gnn_fc_kernel(
    const float* __restrict__ xbar, const float* __restrict__ Sv,
    const float* __restrict__ wg1, const float* __restrict__ bg1,
    const float* __restrict__ wg2, float* __restrict__ gvec) {
  const int b = blockIdx.x;
  __shared__ float xls[Dsz];
  __shared__ float h1[FFsz];
  for (int d = threadIdx.x; d < Dsz; d += 256) xls[d] = xbar[(size_t)b*Dsz + d];
  __syncthreads();
  const int n = threadIdx.x;
  float acc = bg1[n];
  for (int d = 0; d < Dsz; ++d) acc += xls[d]*wg1[(size_t)d*FFsz + n];
  h1[n] = fmaxf(acc, 0.f);
  __syncthreads();
  if (n < 64) {
    float gacc = 0.f;
    for (int f = 0; f < FFsz; ++f) gacc += h1[f]*wg2[f*64 + n];
    gvec[b*64 + n] = (Sv[b] > 0.f) ? gacc : 0.f;
  }
}

// cosine similarity partials per batch
__global__ __launch_bounds__(256) void sim_kernel(
    const float* __restrict__ embs, const float* __restrict__ gvec,
    const float* __restrict__ bg2, const int* __restrict__ mask,
    float* __restrict__ simp) {
  const int b = blockIdx.x;
  __shared__ float gl[64], b2l[64];
  __shared__ float red[256];
  if (threadIdx.x < 64) { gl[threadIdx.x] = gvec[b*64 + threadIdx.x]; b2l[threadIdx.x] = bg2[threadIdx.x]; }
  __syncthreads();
  float cosv = 0.f;
  const int l = threadIdx.x;
  if (l < Lsz) {
    const float* e = embs + ((size_t)b*Lsz + l)*64;
    float mf = mask[b*Lsz + l] ? 1.f : 0.f;
    float num = 0.f, na = 0.f, nb = 0.f;
    for (int c = 0; c < 64; ++c) {
      float ev = e[c];
      float gv = mf*gl[c] + b2l[c];
      num += ev*gv; na += ev*ev; nb += gv*gv;
    }
    na = fmaxf(sqrtf(na), 1e-8f);
    nb = fmaxf(sqrtf(nb), 1e-8f);
    cosv = num/(na*nb);
  }
  red[threadIdx.x] = cosv; __syncthreads();
  for (int o = 128; o > 0; o >>= 1) {
    if (threadIdx.x < o) red[threadIdx.x] += red[threadIdx.x + o];
    __syncthreads();
  }
  if (threadIdx.x == 0) simp[b] = red[0]*(1.f/201.f);
}

__global__ __launch_bounds__(256) void sim_reduce_kernel(
    const float* __restrict__ simp, float* __restrict__ outscalar) {
  __shared__ float red[256];
  red[threadIdx.x] = simp[threadIdx.x];
  __syncthreads();
  for (int o = 128; o > 0; o >>= 1) {
    if (threadIdx.x < o) red[threadIdx.x] += red[threadIdx.x + o];
    __syncthreads();
  }
  if (threadIdx.x == 0) *outscalar = red[0]*(1.f/256.f);
}

// ---------------- output packing ----------------
__global__ __launch_bounds__(256) void pack_kernel(
    const float* __restrict__ embs_pos, const float* __restrict__ embs_neg,
    const float* __restrict__ prec, float* __restrict__ out) {
  const int idx = blockIdx.x*4 + (threadIdx.x >> 6);   // row in [0, BL)
  const int c = threadIdx.x & 63;
  const int b = idx / Lsz, l = idx % Lsz;
  float ep = embs_pos[(size_t)idx*64 + c];
  float pv = prec[(size_t)idx*64 + c];
  float en = embs_neg[(size_t)idx*64 + c];
  if (l < 200) {
    out[O0 + ((size_t)(b*200 + l))*64 + c] = pv;
    out[O2 + ((size_t)(b*200 + l))*64 + c] = en;
  }
  if (l >= 1) out[O1 + ((size_t)(b*200 + l - 1))*64 + c] = ep;
  out[O3 + (size_t)idx*128 + c]      = ep;
  out[O3 + (size_t)idx*128 + 64 + c] = pv;
}

// ---------------- host ----------------
extern "C" void kernel_launch(void* const* d_in, const int* in_sizes, int n_in,
                              void* d_out, int out_size, void* d_ws, size_t ws_size,
                              hipStream_t stream) {
  (void)in_sizes; (void)n_in; (void)out_size; (void)ws_size;
  const float* il    = (const float*)d_in[0];
  const float* nl    = (const float*)d_in[1];
  const float* fc_w  = (const float*)d_in[2];
  const float* fc_b  = (const float*)d_in[3];
  const float* mlp_w = (const float*)d_in[4];
  const float* mlp_b = (const float*)d_in[5];
  const float* pos   = (const float*)d_in[6];
  const float* ln0g  = (const float*)d_in[7];
  const float* ln0b  = (const float*)d_in[8];
  const float* wqkv  = (const float*)d_in[9];
  const float* bqkv  = (const float*)d_in[10];
  const float* wo    = (const float*)d_in[11];
  const float* bo    = (const float*)d_in[12];
  const float* ln1g  = (const float*)d_in[13];
  const float* ln1b  = (const float*)d_in[14];
  const float* w1    = (const float*)d_in[15];
  const float* b1    = (const float*)d_in[16];
  const float* w2    = (const float*)d_in[17];
  const float* b2    = (const float*)d_in[18];
  const float* ln2g  = (const float*)d_in[19];
  const float* ln2b  = (const float*)d_in[20];
  const float* wg1   = (const float*)d_in[21];
  const float* bg1   = (const float*)d_in[22];
  const float* wg2   = (const float*)d_in[23];
  const float* bg2   = (const float*)d_in[24];
  const int*   msk   = (const int*)d_in[25];
  float* dout = (float*)d_out;

  size_t off = 0;
  auto alloc = [&](size_t bytes) -> void* {
    void* p = (char*)d_ws + off;
    off += (bytes + 255) & ~(size_t)255;
    return p;
  };
  bf16_t* fcwT  = (bf16_t*)alloc((size_t)Hsz*Dsz*2);        // 64x1536
  bf16_t* mlpwT = (bf16_t*)alloc((size_t)Hsz*Hsz*2);        // 64x64
  bf16_t* qkvT  = (bf16_t*)alloc((size_t)2*192*Hsz*2);      // 2 x 192x64
  bf16_t* woT   = (bf16_t*)alloc((size_t)2*Hsz*Hsz*2);      // 2 x 64x64
  bf16_t* w1T   = (bf16_t*)alloc((size_t)2*FFsz*Hsz*2);     // 2 x 256x64
  bf16_t* w2T   = (bf16_t*)alloc((size_t)2*Hsz*FFsz*2);     // 2 x 64x256
  float*  embsP = (float*)alloc((size_t)BL*64*4);
  float*  embsN = (float*)alloc((size_t)BL*64*4);
  float*  xcur  = (float*)alloc((size_t)BL*64*4);
  bf16_t* qkv16 = (bf16_t*)alloc((size_t)BL*192*2);         // bf16 qkv buffer
  float*  attnb = (float*)alloc((size_t)BL*64*4);
  float*  tmp1  = (float*)alloc((size_t)BL*256*4);
  float*  xbar  = (float*)alloc((size_t)Bsz*Dsz*4);
  float*  Sv    = (float*)alloc((size_t)Bsz*4);
  float*  gvec  = (float*)alloc((size_t)Bsz*64*4);
  float*  simp  = (float*)alloc((size_t)Bsz*4);

  dim3 blk(256);
  // weights -> bf16, transposed to [N][K]
  cvt_transpose_kernel<<<(Dsz*Hsz + 255)/256, blk, 0, stream>>>(fc_w, fcwT, Dsz, Hsz);
  cvt_transpose_kernel<<<(Hsz*Hsz + 255)/256, blk, 0, stream>>>(mlp_w, mlpwT, Hsz, Hsz);
  for (int i = 0; i < 2; ++i) {
    cvt_transpose_kernel<<<(Hsz*192 + 255)/256, blk, 0, stream>>>(wqkv + i*Hsz*192, qkvT + i*192*Hsz, Hsz, 192);
    cvt_transpose_kernel<<<(Hsz*Hsz + 255)/256, blk, 0, stream>>>(wo + i*Hsz*Hsz, woT + i*Hsz*Hsz, Hsz, Hsz);
    cvt_transpose_kernel<<<(Hsz*FFsz + 255)/256, blk, 0, stream>>>(w1 + i*Hsz*FFsz, w1T + i*FFsz*Hsz, Hsz, FFsz);
    cvt_transpose_kernel<<<(FFsz*Hsz + 255)/256, blk, 0, stream>>>(w2 + i*FFsz*Hsz, w2T + i*Hsz*FFsz, FFsz, Hsz);
  }

  // embeddings (dominant WMMA GEMMs, K=1536 fused with K=64 + gelu)
  fc_mlp_kernel<<<402, blk, 0, stream>>>(il, fcwT, fc_b, mlpwT, mlp_b, embsP);
  fc_mlp_kernel<<<402, blk, 0, stream>>>(nl, fcwT, fc_b, mlpwT, mlp_b, embsN);

  // GNN (rank-1 collapse) + similarity
  gnn_mean_kernel<<<Bsz, blk, 0, stream>>>(il, msk, xbar, Sv);
  gnn_fc_kernel<<<Bsz, blk, 0, stream>>>(xbar, Sv, wg1, bg1, wg2, gvec);
  sim_kernel<<<Bsz, blk, 0, stream>>>(embsP, gvec, bg2, msk, simp);
  sim_reduce_kernel<<<1, blk, 0, stream>>>(simp, dout + OSIM);

  // transformer encoder
  resid_ln_kernel<<<BL/8, blk, 0, stream>>>(embsP, pos, 0, ln0g, ln0b, xcur);
  for (int i = 0; i < 2; ++i) {
    gemm_wmma_bf16out_kernel<64><<<dim3(402, 3), blk, 0, stream>>>(xcur, qkvT + i*192*Hsz, bqkv + i*192, qkv16, 192);
    attn_kernel<<<Bsz*2, blk, 0, stream>>>(qkv16, msk, attnb);
    gemm_wmma_kernel<64, 0><<<dim3(402, 1), blk, 0, stream>>>(attnb, woT + i*Hsz*Hsz, bo + i*64, tmp1, 64);
    resid_ln_kernel<<<BL/8, blk, 0, stream>>>(xcur, tmp1, 1, ln1g + i*64, ln1b + i*64, xcur);
    gemm_wmma_kernel<64, 1><<<dim3(402, 4), blk, 0, stream>>>(xcur, w1T + i*FFsz*Hsz, b1 + i*FFsz, tmp1, 256);
    gemm_wmma_kernel<256, 0><<<dim3(402, 1), blk, 0, stream>>>(tmp1, w2T + i*Hsz*FFsz, b2 + i*64, attnb, 64);
    resid_ln_kernel<<<BL/8, blk, 0, stream>>>(xcur, attnb, 1, ln2g + i*64, ln2b + i*64, xcur);
  }

  // outputs
  pack_kernel<<<BL/4, blk, 0, stream>>>(embsP, embsN, xcur, dout);
}
